// PrismEncoder_82497731821637
// MI455X (gfx1250) — matrix operand order repaired and checked by reading, not compiled, not run on hardware
//
#include <hip/hip_runtime.h>
#include <hip/hip_bf16.h>
#include <stdint.h>

// ---------------- problem constants (match reference) ----------------
constexpr int CB   = 128;   // batch
constexpr int CN1  = 256;   // nodes (x1)
constexpr int CN2  = 256;   // nodes (x2)
constexpr int CD   = 256;   // model dim
constexpr int CH   = 8;     // heads
constexpr int CDH  = 32;    // head dim
constexpr int CDFF = 1024;  // ffn hidden
constexpr int CM   = CB * CN1; // 32768 token rows

typedef _Float16 v16h __attribute__((ext_vector_type(16)));
typedef _Float16 v8h  __attribute__((ext_vector_type(8)));
typedef float    v8f  __attribute__((ext_vector_type(8)));

// Build a v16h fragment from two 8-halfword chunks (each lowers to b128 load)
__device__ __forceinline__ v16h frag2(const _Float16* p0, const _Float16* p1) {
    v8h lo = *(const v8h*)p0;
    v8h hi = *(const v8h*)p1;
    v16h r;
#pragma unroll
    for (int i = 0; i < 8; ++i) { r[i] = lo[i]; r[i + 8] = hi[i]; }
    return r;
}

__device__ __forceinline__ v8f wmma32(v16h a, v16h b, v8f c) {
    // D = A(16x32 f16) * B(32x16 f16) + C(16x16 f32)
    return __builtin_amdgcn_wmma_f32_16x16x32_f16(false, a, false, b, (short)0, c,
                                                  false, false);
}

// ---------------- conversion kernels ----------------
__global__ void k_f32_to_f16(const float* __restrict__ src,
                             _Float16* __restrict__ dst, int n) {
    int i = blockIdx.x * blockDim.x + threadIdx.x;
    if (i < n) dst[i] = (_Float16)src[i];
}

// W[k][n] (row-major KxN) -> Wt[n][k] f16 (row-major NxK)
__global__ void k_transpose_f16(const float* __restrict__ src,
                                _Float16* __restrict__ dst, int K, int N) {
    int i = blockIdx.x * blockDim.x + threadIdx.x;
    if (i < K * N) {
        int k = i / N, n = i - k * N;
        dst[(size_t)n * K + k] = (_Float16)src[i];
    }
}

// ---------------- generic WMMA GEMM ----------------
// C[M,N] = A[M,K](f16,row-major) * Bt[N,K](f16, pre-transposed) + bias
// optional: f32 residual add, relu, f32 output, f16 output (row-major or
// v-transposed [b][h][d][tok] for attention V operands).
__global__ void __launch_bounds__(256)
k_gemm_wmma(const _Float16* __restrict__ A, int lda,
            const _Float16* __restrict__ Bt,
            const float* __restrict__ bias,
            int M, int N, int K,
            const float* __restrict__ resid,
            float* __restrict__ outF,
            _Float16* __restrict__ outH,
            int vtrans, int relu,
            int ntok, int nheads, int dh) {
    const int wave = blockIdx.x * (blockDim.x >> 5) + (threadIdx.x >> 5);
    const int lane = threadIdx.x & 31;
    const int tilesN = N >> 5;
    const int tm = wave / tilesN, tn = wave - tm * tilesN;
    const int m0 = tm << 5, n0 = tn << 5;
    if (m0 >= M) return;
    const int mr = lane & 15, half = lane >> 4;

    v8f acc[2][2] = {{{}, {}}, {{}, {}}};

    const _Float16* a0p = A + (size_t)(m0 + mr) * lda + half * 16;
    const _Float16* a1p = A + (size_t)(m0 + 16 + mr) * lda + half * 16;
    const _Float16* b0p = Bt + (size_t)(n0 + mr) * K + half * 16;
    const _Float16* b1p = Bt + (size_t)(n0 + 16 + mr) * K + half * 16;

    for (int k0 = 0; k0 < K; k0 += 32) {
        __builtin_prefetch(a0p + k0 + 128, 0, 1);
        v16h a0 = frag2(a0p + k0, a0p + k0 + 8);
        v16h a1 = frag2(a1p + k0, a1p + k0 + 8);
        v16h b0 = frag2(b0p + k0, b0p + k0 + 8);
        v16h b1 = frag2(b1p + k0, b1p + k0 + 8);
        acc[0][0] = wmma32(a0, b0, acc[0][0]);
        acc[0][1] = wmma32(a0, b1, acc[0][1]);
        acc[1][0] = wmma32(a1, b0, acc[1][0]);
        acc[1][1] = wmma32(a1, b1, acc[1][1]);
    }

#pragma unroll
    for (int mi = 0; mi < 2; ++mi) {
#pragma unroll
        for (int ni = 0; ni < 2; ++ni) {
            const int n = n0 + ni * 16 + mr;
            const float bv = bias ? bias[n] : 0.0f;
#pragma unroll
            for (int r = 0; r < 8; ++r) {
                const int row = m0 + mi * 16 + r + half * 8;
                float val = acc[mi][ni][r] + bv;
                if (resid) val += resid[(size_t)row * N + n];
                if (relu)  val = fmaxf(val, 0.0f);
                if (outF)  outF[(size_t)row * N + n] = val;
                if (outH) {
                    if (!vtrans) {
                        outH[(size_t)row * N + n] = (_Float16)val;
                    } else {
                        const int bb  = row / ntok, tok = row - bb * ntok;
                        const int hh  = n / dh,     dd  = n - hh * dh;
                        outH[(((size_t)bb * nheads + hh) * dh + dd) * (size_t)ntok + tok] =
                            (_Float16)val;
                    }
                }
            }
        }
    }
}

// ---------------- fused dual attention ----------------
// One wave handles one (b, h, 16-query tile); self-attn (adj-masked) + cross.
// LDS per wave: Sbuf 16x256 f32 (16KB) + Pbuf 16x256 f16 (8KB) = 24KB.
__global__ void __launch_bounds__(128)
k_attention(const _Float16* __restrict__ qh,
            const _Float16* __restrict__ k1h, const _Float16* __restrict__ v1T,
            const _Float16* __restrict__ k2h, const _Float16* __restrict__ v2T,
            const float* __restrict__ adj1,  const float* __restrict__ mask1,
            const float* __restrict__ mask2, _Float16* __restrict__ ch) {
    extern __shared__ char smem[];
    const int wv = threadIdx.x >> 5, lane = threadIdx.x & 31;
    float*    Sbuf = (float*)(smem + wv * 24576);
    _Float16* Pbuf = (_Float16*)(smem + wv * 24576 + 16384);

    const int gw = blockIdx.x * 4 + wv;
    const int qt = gw & 15, h = (gw >> 4) & 7, b = gw >> 7;
    const int q0 = qt * 16;
    const int mr = lane & 15, half = lane >> 4;
    const float scale = 0.17677669529663687f; // 1/sqrt(32)

    // Q fragment (A 16x32): rows = queries, K = head dim
    const _Float16* qb = qh + (((size_t)b * CN1 + q0 + mr) * CH + h) * CDH + half * 16;
    const v16h aq = frag2(qb, qb + 8);

    v8f acc0 = {}, acc1 = {};

#pragma unroll 1
    for (int phase = 0; phase < 2; ++phase) {
        const _Float16* kh = phase ? k2h : k1h;
        const _Float16* vT = phase ? v2T : v1T;

        // ---- scores: 16 key tiles, one WMMA each (K = DH = 32) ----
        const int key = (lane & 15); // column within tile; key = kt*16 + mr
        const float mcol = phase ? mask2[(size_t)b * CN2 + 0] : 0.0f; (void)mcol;
#pragma unroll 1
        for (int kt = 0; kt < 16; ++kt) {
            const _Float16* kb =
                kh + (((size_t)b * CN1 + kt * 16 + mr) * CH + h) * CDH + half * 16;
            v16h bk = frag2(kb, kb + 8);
            v8f z = {};
            v8f s = wmma32(aq, bk, z);
            const int kcol = kt * 16 + key;
            const float mc = phase ? mask2[(size_t)b * CN2 + kcol]
                                   : mask1[(size_t)b * CN1 + kcol];
#pragma unroll
            for (int r = 0; r < 8; ++r) {
                const int m = r + half * 8;
                float val;
                if (phase == 0) {
                    const float a = adj1[((size_t)b * CN1 + (q0 + m)) * CN1 + kcol];
                    val = (a * mc > 0.0f) ? s[r] * scale : -1e9f;
                } else {
                    val = (mc > 0.0f) ? s[r] * scale : -1e9f;
                }
                Sbuf[m * 256 + kcol] = val;
            }
        }
        asm volatile("s_wait_dscnt 0" ::: "memory");

        // ---- row softmax (lane pair {l, l+16} covers one row) ----
        const int srow = mr, cb = half * 128;
        float mx = -3.4e38f;
        for (int c = 0; c < 128; ++c) mx = fmaxf(mx, Sbuf[srow * 256 + cb + c]);
        mx = fmaxf(mx, __shfl_xor(mx, 16, 32));
        float sum = 0.0f;
        for (int c = 0; c < 128; ++c) {
            const float e = __expf(Sbuf[srow * 256 + cb + c] - mx);
            sum += e;
            Pbuf[srow * 256 + cb + c] = (_Float16)e;
        }
        sum += __shfl_xor(sum, 16, 32);
        const float inv = 1.0f / sum;
        asm volatile("s_wait_dscnt 0" ::: "memory");
        for (int c = 0; c < 128; ++c)
            Pbuf[srow * 256 + cb + c] =
                (_Float16)((float)Pbuf[srow * 256 + cb + c] * inv);
        asm volatile("s_wait_dscnt 0" ::: "memory");

        // ---- PV: 8 iterations of 32 keys; 2 WMMAs each (DH = 32 cols) ----
        const _Float16* vb0 =
            vT + (((size_t)b * CH + h) * CDH + mr) * (size_t)CN1 + half * 16;
        const _Float16* vb1 =
            vT + (((size_t)b * CH + h) * CDH + 16 + mr) * (size_t)CN1 + half * 16;
#pragma unroll 1
        for (int kt2 = 0; kt2 < 8; ++kt2) {
            const _Float16* pp = Pbuf + mr * 256 + kt2 * 32 + half * 16;
            v16h ap  = frag2(pp, pp + 8);
            v16h bv0 = frag2(vb0 + kt2 * 32, vb0 + kt2 * 32 + 8);
            v16h bv1 = frag2(vb1 + kt2 * 32, vb1 + kt2 * 32 + 8);
            acc0 = wmma32(ap, bv0, acc0);
            acc1 = wmma32(ap, bv1, acc1);
        }
    }

    // ---- write context (c1 + c2) as f16 row-major [b][tok][h][d] ----
#pragma unroll
    for (int r = 0; r < 8; ++r) {
        const int m = r + half * 8;
        const size_t base = (((size_t)b * CN1 + (q0 + m)) * CH + h) * CDH;
        ch[base + mr]      = (_Float16)acc0[r];
        ch[base + 16 + mr] = (_Float16)acc1[r];
    }
}

// ---------------- layernorm (one row of D=256 per block) ----------------
__global__ void __launch_bounds__(256)
k_layernorm(const float* __restrict__ x, const float* __restrict__ g,
            const float* __restrict__ be, float* __restrict__ outF,
            _Float16* __restrict__ outH) {
    __shared__ float red[256];
    const int row = blockIdx.x, t = threadIdx.x;
    const float v = x[(size_t)row * CD + t];
    red[t] = v;
    __syncthreads();
    for (int s = 128; s > 0; s >>= 1) { if (t < s) red[t] += red[t + s]; __syncthreads(); }
    const float mu = red[0] * (1.0f / CD);
    __syncthreads();
    const float d = v - mu;
    red[t] = d * d;
    __syncthreads();
    for (int s = 128; s > 0; s >>= 1) { if (t < s) red[t] += red[t + s]; __syncthreads(); }
    const float var = red[0] * (1.0f / CD);
    const float o = g[t] * d * rsqrtf(var + 1e-6f) + be[t];
    outF[(size_t)row * CD + t] = o;
    if (outH) outH[(size_t)row * CD + t] = (_Float16)o;
}

// ---------------- host orchestration ----------------
extern "C" void kernel_launch(void* const* d_in, const int* in_sizes, int n_in,
                              void* d_out, int out_size, void* d_ws, size_t ws_size,
                              hipStream_t stream) {
    (void)in_sizes; (void)n_in; (void)out_size; (void)ws_size;
    const float* x1    = (const float*)d_in[0];
    const float* adj1  = (const float*)d_in[1];
    const float* mask1 = (const float*)d_in[2];
    const float* x2    = (const float*)d_in[3];
    const float* mask2 = (const float*)d_in[4];
    const float* Wq = (const float*)d_in[5];  const float* bq = (const float*)d_in[6];
    const float* Wk = (const float*)d_in[7];  const float* bk = (const float*)d_in[8];
    const float* Wv = (const float*)d_in[9];  const float* bv = (const float*)d_in[10];
    const float* Wo = (const float*)d_in[11]; const float* bo = (const float*)d_in[12];
    const float* W1 = (const float*)d_in[13]; const float* b1 = (const float*)d_in[14];
    const float* W2 = (const float*)d_in[15]; const float* b2 = (const float*)d_in[16];
    const float* g1 = (const float*)d_in[17]; const float* be1 = (const float*)d_in[18];
    const float* g2 = (const float*)d_in[19]; const float* be2 = (const float*)d_in[20];
    float* out = (float*)d_out;

    char* ws = (char*)d_ws;
    const size_t SZH = (size_t)CM * CD * 2;  // 16 MiB (f16 [M][256])
    const size_t SZF = (size_t)CM * CD * 4;  // 32 MiB (f32 [M][256])
    _Float16* x1h  = (_Float16*)(ws + 0 * SZH);          // later reused: out1h
    _Float16* x2h  = (_Float16*)(ws + 1 * SZH);          // later reused: ch
    _Float16* qh   = (_Float16*)(ws + 2 * SZH);          // later reused: hidh
    _Float16* k1h  = (_Float16*)(ws + 3 * SZH);
    _Float16* v1T  = (_Float16*)(ws + 4 * SZH);
    _Float16* k2h  = (_Float16*)(ws + 5 * SZH);
    _Float16* v2T  = (_Float16*)(ws + 6 * SZH);
    float*    pre  = (float*)(ws + 7 * SZH);             // pre-LN buffer (reused)
    float*    out1 = (float*)(ws + 7 * SZH + SZF);
    char*     wbase = ws + 7 * SZH + 2 * SZF;
    _Float16* WqT = (_Float16*)(wbase + 0 * 131072);
    _Float16* WkT = (_Float16*)(wbase + 1 * 131072);
    _Float16* WvT = (_Float16*)(wbase + 2 * 131072);
    _Float16* WoT = (_Float16*)(wbase + 3 * 131072);
    _Float16* W1T = (_Float16*)(wbase + 4 * 131072);
    _Float16* W2T = (_Float16*)(wbase + 4 * 131072 + 524288);
    _Float16* ch    = x2h;                                // alias (x2h consumed)
    _Float16* out1h = x1h;                                // alias (x1h consumed)
    _Float16* hidh  = qh;                                 // alias (qh.. consumed)

    const int nXD = CM * CD;
    // 1) convert inputs / weights to f16 (weights transposed to [N][K])
    hipLaunchKernelGGL(k_f32_to_f16, dim3(nXD / 256), dim3(256), 0, stream, x1, x1h, nXD);
    hipLaunchKernelGGL(k_f32_to_f16, dim3(nXD / 256), dim3(256), 0, stream, x2, x2h, nXD);
    hipLaunchKernelGGL(k_transpose_f16, dim3(256),  dim3(256), 0, stream, Wq, WqT, CD, CD);
    hipLaunchKernelGGL(k_transpose_f16, dim3(256),  dim3(256), 0, stream, Wk, WkT, CD, CD);
    hipLaunchKernelGGL(k_transpose_f16, dim3(256),  dim3(256), 0, stream, Wv, WvT, CD, CD);
    hipLaunchKernelGGL(k_transpose_f16, dim3(256),  dim3(256), 0, stream, Wo, WoT, CD, CD);
    hipLaunchKernelGGL(k_transpose_f16, dim3(1024), dim3(256), 0, stream, W1, W1T, CD, CDFF);
    hipLaunchKernelGGL(k_transpose_f16, dim3(1024), dim3(256), 0, stream, W2, W2T, CDFF, CD);

    // 2) projections: waves = (M/32)*(N/32); 8 waves / 256-thread block
    const dim3 gP((CM / 32) * (CD / 32) / 8), tB(256);
    hipLaunchKernelGGL(k_gemm_wmma, gP, tB, 0, stream, x1h, CD, WqT, bq, CM, CD, CD,
                       (const float*)nullptr, (float*)nullptr, qh, 0, 0, CN1, CH, CDH);
    hipLaunchKernelGGL(k_gemm_wmma, gP, tB, 0, stream, x1h, CD, WkT, bk, CM, CD, CD,
                       (const float*)nullptr, (float*)nullptr, k1h, 0, 0, CN1, CH, CDH);
    hipLaunchKernelGGL(k_gemm_wmma, gP, tB, 0, stream, x1h, CD, WvT, bv, CM, CD, CD,
                       (const float*)nullptr, (float*)nullptr, v1T, 1, 0, CN1, CH, CDH);
    hipLaunchKernelGGL(k_gemm_wmma, gP, tB, 0, stream, x2h, CD, WkT, bk, CM, CD, CD,
                       (const float*)nullptr, (float*)nullptr, k2h, 0, 0, CN2, CH, CDH);
    hipLaunchKernelGGL(k_gemm_wmma, gP, tB, 0, stream, x2h, CD, WvT, bv, CM, CD, CD,
                       (const float*)nullptr, (float*)nullptr, v2T, 1, 0, CN2, CH, CDH);

    // 3) fused dual attention: B*H*(N1/16) waves, 4 waves/block, 96KB LDS
    hipLaunchKernelGGL(k_attention, dim3(CB * CH * (CN1 / 16) / 4), dim3(128),
                       4 * 24576, stream, qh, k1h, v1T, k2h, v2T, adj1, mask1, mask2, ch);

    // 4) attention output projection + residual(x1) -> pre ; LN1 -> out1(+f16)
    hipLaunchKernelGGL(k_gemm_wmma, gP, tB, 0, stream, ch, CD, WoT, bo, CM, CD, CD,
                       x1, pre, (_Float16*)nullptr, 0, 0, CN1, CH, CDH);
    hipLaunchKernelGGL(k_layernorm, dim3(CM), dim3(256), 0, stream, pre, g1, be1,
                       out1, out1h);

    // 5) FFN: relu(out1h @ W1 + b1) -> hidh ; hidh @ W2 + b2 + out1 -> pre ; LN2
    const dim3 gF1((CM / 32) * (CDFF / 32) / 8);
    hipLaunchKernelGGL(k_gemm_wmma, gF1, tB, 0, stream, out1h, CD, W1T, b1, CM, CDFF, CD,
                       (const float*)nullptr, (float*)nullptr, hidh, 0, 1, CN1, CH, CDH);
    hipLaunchKernelGGL(k_gemm_wmma, gP, tB, 0, stream, hidh, CDFF, W2T, b2, CM, CD, CDFF,
                       out1, pre, (_Float16*)nullptr, 0, 0, CN1, CH, CDH);
    hipLaunchKernelGGL(k_layernorm, dim3(CM), dim3(256), 0, stream, pre, g2, be2,
                       out, (_Float16*)nullptr);
}
